// WeaklySupervisedDetection_76132590289199
// MI455X (gfx1250) — matrix operand order repaired and checked by reading, not compiled, not run on hardware
//
#include <hip/hip_runtime.h>
#include <hip/hip_bf16.h>
#include <math.h>

#define Hh 50
#define Ww 50
#define Cc 512
#define Aa 9
#define NCc 20
#define Kk 2048
#define DINd 4608         // 3*3*512
#define NPIX 2500         // 50*50
#define NANCH 22500       // 2500*9
#define MCONV 2560        // conv GEMM rows padded to 256
#define KCONV 2048        // 2*2*512
#define NCONV 512
#define NFC6 9216
#define NFC7 4096
#define LDSS 40           // padded LDS row stride (bf16 elems) = 80B

typedef __attribute__((ext_vector_type(16))) __bf16 v16bf;
typedef __attribute__((ext_vector_type(8)))  float  v8f;

struct __align__(16) U4 { unsigned int x, y, z, w; };
struct U8x { U4 a, b; };

// ---- gfx1250 async global->LDS path (feature-guarded; falls back to sync) ----
#if defined(__has_builtin)
# if __has_builtin(__builtin_amdgcn_global_load_async_to_lds_b128) && \
     __has_builtin(__builtin_amdgcn_s_wait_asynccnt)
#  define USE_ASYNC_LDS 1
# endif
#endif
#ifndef USE_ASYNC_LDS
# define USE_ASYNC_LDS 0
#endif

#if USE_ASYNC_LDS
typedef int v4i_async __attribute__((vector_size(4 * sizeof(int))));
#endif

__device__ __forceinline__ unsigned short f2bf(float f) {
  unsigned int u = __float_as_uint(f);
  unsigned int r = u + 0x7FFFu + ((u >> 16) & 1u);   // round-to-nearest-even
  return (unsigned short)(r >> 16);
}
__device__ __forceinline__ float sigmoidf_(float x) { return 1.0f / (1.0f + __expf(-x)); }
__device__ __forceinline__ int imin_(int a, int b) { return a < b ? a : b; }
__device__ __forceinline__ int imax_(int a, int b) { return a > b ? a : b; }

// ---------------------------------------------------------------------------
// bf16 WMMA GEMM: C[M,N] = relu( A_bf16[M,K] @ B_f32[K,N] + bias )
// Block tile 256x128, 8 waves (4Mx2N), wave tile 64x64 (4x4 fragments),
// K-step 32, double-buffered LDS, async global->LDS for the A tile.
// WMODE: 0 -> write f32, 1 -> write bf16.
// ---------------------------------------------------------------------------
template <int WMODE>
__global__ __launch_bounds__(256)
void wmma_gemm(const unsigned short* __restrict__ A,   // [M,K] bf16 bits
               const float* __restrict__ B,            // [K,N] f32
               const float* __restrict__ bias,         // [N]
               void* __restrict__ Cout,                // [M,N] f32 or bf16
               int M, int N, int K)
{
  __shared__ __align__(16) unsigned short As[2][256 * LDSS];  // [row][k] row-major
  __shared__ __align__(16) unsigned short Bs[2][128 * LDSS];  // [n][k]   n-major

  const int tid   = threadIdx.x;
  const int wave  = tid >> 5;
  const int lane  = tid & 31;
  const int waveM = wave >> 1;        // 0..3 -> M offset *64
  const int waveN = wave & 1;         // 0..1 -> N offset *64
  const int m16   = lane & 15;
  const int kh    = lane >> 4;        // K-half per ISA 16-bit layouts

  const int nBase = blockIdx.x * 128;
  const int mBase = blockIdx.y * 256;

  v8f acc[4][4] = {};

  const int stB_n  = tid & 127;
  const int stB_kh = tid >> 7;        // 0/1 -> k 0..15 / 16..31

  const int nk = K >> 5;

  // ---- tile staging (A: async copy if available; B: fp32->bf16 transpose) ----
  auto stage = [&](int kt, int buf) {
    const int k0 = kt << 5;
    // A tile: 256 rows x 2 16B granules (k 0..7, 16..23) = 512; 2 per thread
#pragma unroll
    for (int g2 = 0; g2 < 2; ++g2) {
      int g    = tid + g2 * 256;
      int row  = g >> 1;
      int half = g & 1;
      const unsigned short* src = A + (size_t)(mBase + row) * K + k0 + half * 16;
      unsigned short*       dst = &As[buf][row * LDSS + half * 16];
#if USE_ASYNC_LDS
      __builtin_amdgcn_global_load_async_to_lds_b128((v4i_async*)src, (v4i_async*)dst, 0, 0);
#else
      *reinterpret_cast<U4*>(dst) = *reinterpret_cast<const U4*>(src);
#endif
    }
    // B tile: 32x128 fp32 -> bf16, transposed to [n][k]
    unsigned int pk[8];
    const float* src = B + (size_t)(k0 + stB_kh * 16) * N + nBase + stB_n;
#pragma unroll
    for (int i = 0; i < 8; ++i) {
      float v0 = src[(size_t)(2 * i) * N];
      float v1 = src[(size_t)(2 * i + 1) * N];
      pk[i] = (unsigned int)f2bf(v0) | ((unsigned int)f2bf(v1) << 16);
    }
    U4* dst = reinterpret_cast<U4*>(&Bs[buf][stB_n * LDSS + stB_kh * 16]);
    U4 d0; d0.x = pk[0]; d0.y = pk[1]; d0.z = pk[2]; d0.w = pk[3];
    U4 d1; d1.x = pk[4]; d1.y = pk[5]; d1.z = pk[6]; d1.w = pk[7];
    dst[0] = d0; dst[1] = d1;
  };

  stage(0, 0);
#if USE_ASYNC_LDS
  __builtin_amdgcn_s_wait_asynccnt(0);
#endif
  __syncthreads();

  for (int kt = 0; kt < nk; ++kt) {
    const int cur = kt & 1;
    if (kt + 1 < nk) stage(kt + 1, cur ^ 1);   // overlap next-tile staging

    // ---- fragment loads per ISA 16-bit A/B layouts ----
    v16bf afr[4], bfr[4];
#pragma unroll
    for (int fm = 0; fm < 4; ++fm) {
      int row = waveM * 64 + fm * 16 + m16;
      U8x t;
      t.a = *reinterpret_cast<const U4*>(&As[cur][row * LDSS + kh * 8]);       // K=kh*8..+8
      t.b = *reinterpret_cast<const U4*>(&As[cur][row * LDSS + 16 + kh * 8]);  // K=16+kh*8..+8
      afr[fm] = __builtin_bit_cast(v16bf, t);
    }
#pragma unroll
    for (int fn = 0; fn < 4; ++fn) {
      int n = waveN * 64 + fn * 16 + m16;
      U8x t;
      t.a = *reinterpret_cast<const U4*>(&Bs[cur][n * LDSS + kh * 8]);         // K=kh*8..+8
      t.b = *reinterpret_cast<const U4*>(&Bs[cur][n * LDSS + 16 + kh * 8]);    // K=16+kh*8..+8
      bfr[fn] = __builtin_bit_cast(v16bf, t);
    }
#pragma unroll
    for (int fm = 0; fm < 4; ++fm)
#pragma unroll
      for (int fn = 0; fn < 4; ++fn)
        acc[fm][fn] = __builtin_amdgcn_wmma_f32_16x16x32_bf16(
            false, afr[fm], false, bfr[fn], (short)0, acc[fm][fn], false, false);

#if USE_ASYNC_LDS
    __builtin_amdgcn_s_wait_asynccnt(0);
#endif
    __syncthreads();
  }

  // ---- epilogue: C layout (VGPR r: lanes0-15 M=r, lanes16-31 M=8+r; N=lane&15) ----
#pragma unroll
  for (int fn = 0; fn < 4; ++fn) {
    int col = nBase + waveN * 64 + fn * 16 + m16;
    float bv = bias[col];
#pragma unroll
    for (int fm = 0; fm < 4; ++fm) {
#pragma unroll
      for (int r = 0; r < 8; ++r) {
        int rowl = waveM * 64 + fm * 16 + kh * 8 + r;
        float v = fmaxf(acc[fm][fn][r] + bv, 0.0f);
        size_t o = (size_t)(mBase + rowl) * N + col;
        if (WMODE == 0) ((float*)Cout)[o] = v;
        else            ((unsigned short*)Cout)[o] = f2bf(v);
      }
    }
  }
}

// ---------------------------------------------------------------------------
// im2col for 2x2 SAME conv (pad lo=0, hi=1): A0[r,k] bf16, r=pixel (pad to 2560)
// k = (dy*2+dx)*512 + c  -> matches flat rpn_w [2,2,512,512] as [2048,512].
// ---------------------------------------------------------------------------
__global__ void im2col_kernel(const float* __restrict__ feat, unsigned short* __restrict__ A0)
{
  size_t gid = (size_t)blockIdx.x * blockDim.x + threadIdx.x;
  if (gid >= (size_t)MCONV * KCONV) return;
  int r = (int)(gid / KCONV);
  int k = (int)(gid % KCONV);
  float v = 0.0f;
  if (r < NPIX) {
    int i = r / Ww, j = r % Ww;
    int dy = k >> 10;
    int dx = (k >> 9) & 1;
    int c  = k & 511;
    int ii = i + dy, jj = j + dx;
    if (ii < Hh && jj < Ww) v = feat[((size_t)ii * Ww + jj) * Cc + c];
  }
  A0[gid] = f2bf(v);
}

// 1x1 convs: obj = sigmoid(x @ cls_w + cls_b) [2500*9], reg flat [2500*36]
__global__ __launch_bounds__(64)
void clsreg_kernel(const float* __restrict__ x, const float* __restrict__ cls_w,
                   const float* __restrict__ cls_b, const float* __restrict__ reg_w,
                   const float* __restrict__ reg_b, float* __restrict__ obj,
                   float* __restrict__ reg4)
{
  __shared__ float xs[Cc];
  int p = blockIdx.x;
  int t = threadIdx.x;
  for (int i = t; i < Cc; i += 64) xs[i] = x[(size_t)p * Cc + i];
  __syncthreads();
  if (t < Aa) {
    float acc = cls_b[t];
    for (int c = 0; c < Cc; ++c) acc += xs[c] * cls_w[c * Aa + t];
    obj[p * Aa + t] = sigmoidf_(acc);
  } else if (t < Aa + 36) {
    int q = t - Aa;
    float acc = reg_b[q];
    for (int c = 0; c < Cc; ++c) acc += xs[c] * reg_w[c * 36 + q];
    reg4[(size_t)p * 36 + q] = acc;   // == reshape(-1,4) layout
  }
}

// Exact top-K by rank counting: rank(e) = #{j : v_j > v_e || (v_j==v_e && j<e)}
__global__ __launch_bounds__(256)
void topk_kernel(const float* __restrict__ obj, int* __restrict__ idx)
{
  __shared__ float tile[256];
  int e = blockIdx.x * 256 + threadIdx.x;
  float myv = (e < NANCH) ? obj[e] : 0.0f;
  int rank = 0;
  for (int t0 = 0; t0 < NANCH; t0 += 256) {
    int j = t0 + threadIdx.x;
    tile[threadIdx.x] = (j < NANCH) ? obj[j] : -3.4e38f;
    __syncthreads();
    int lim = imin_(256, NANCH - t0);
    if (e < NANCH) {
      for (int jj = 0; jj < lim; ++jj) {
        float v = tile[jj];
        int j2 = t0 + jj;
        rank += ((v > myv) || (v == myv && j2 < e)) ? 1 : 0;
      }
    }
    __syncthreads();
  }
  if (e < NANCH && rank < Kk) idx[rank] = e;
}

__global__ void boxes_kernel(const int* __restrict__ idx, const float* __restrict__ reg4,
                             float* __restrict__ boxes, float* __restrict__ out_boxes)
{
  int k = blockIdx.x * blockDim.x + threadIdx.x;
  if (k >= Kk) return;
  int e = idx[k];
  float s0 = sigmoidf_(reg4[(size_t)e * 4 + 0]);
  float s1 = sigmoidf_(reg4[(size_t)e * 4 + 1]);
  float s2 = sigmoidf_(reg4[(size_t)e * 4 + 2]);
  float s3 = sigmoidf_(reg4[(size_t)e * 4 + 3]);
  float x1 = s0 * ((float)Ww - 2.0f);
  float x2 = x1 + 1.0f + s1 * ((float)Ww - 1.0f - x1);
  float y1 = s2 * ((float)Hh - 2.0f);
  float y2 = y1 + 1.0f + s3 * ((float)Hh - 1.0f - y1);
  boxes[k * 4 + 0] = x1; boxes[k * 4 + 1] = x2;
  boxes[k * 4 + 2] = y1; boxes[k * 4 + 3] = y2;
  out_boxes[k * 4 + 0] = x1; out_boxes[k * 4 + 1] = x2;
  out_boxes[k * 4 + 2] = y1; out_boxes[k * 4 + 3] = y2;
}

// ROI-align (3x3 bins, 1 sample/bin) -> pooled bf16 [K, 4608]
__global__ __launch_bounds__(128)
void roialign_kernel(const float* __restrict__ feat, const float* __restrict__ boxes,
                     unsigned short* __restrict__ pooled)
{
  int cell = blockIdx.x;              // 0..8 = py*3+px
  int k = blockIdx.y;
  int py = cell / 3, px = cell % 3;
  float x1 = boxes[k * 4 + 0], x2 = boxes[k * 4 + 1];
  float y1 = boxes[k * 4 + 2], y2 = boxes[k * 4 + 3];
  float tx = (px + 0.5f) / 3.0f;
  float ty = (py + 0.5f) / 3.0f;
  float xs = x1 + (x2 - x1) * tx;
  float ys = y1 + (y2 - y1) * ty;
  float x0f = floorf(xs), y0f = floorf(ys);
  float fx = xs - x0f, fy = ys - y0f;
  int x0  = imin_(imax_((int)x0f, 0), Ww - 1);
  int x1i = imin_(x0 + 1, Ww - 1);
  int y0  = imin_(imax_((int)y0f, 0), Hh - 1);
  int y1i = imin_(y0 + 1, Hh - 1);
  float w00 = (1 - fy) * (1 - fx), w01 = (1 - fy) * fx;
  float w10 = fy * (1 - fx),       w11 = fy * fx;
  const float* p00 = feat + ((size_t)y0 * Ww + x0) * Cc;
  const float* p01 = feat + ((size_t)y0 * Ww + x1i) * Cc;
  const float* p10 = feat + ((size_t)y1i * Ww + x0) * Cc;
  const float* p11 = feat + ((size_t)y1i * Ww + x1i) * Cc;
  unsigned short* dst = pooled + (size_t)k * DINd + cell * Cc;
  for (int c = threadIdx.x; c < Cc; c += 128) {
    float v = p00[c] * w00 + p01[c] * w01 + p10[c] * w10 + p11[c] * w11;
    dst[c] = f2bf(v);
  }
}

// fc8c/fc8d logits: [K,20] each
__global__ __launch_bounds__(256)
void fc8_kernel(const float* __restrict__ h7, const float* __restrict__ wc,
                const float* __restrict__ bc, const float* __restrict__ wd,
                const float* __restrict__ bd, float* __restrict__ cl, float* __restrict__ dl)
{
  __shared__ float hrow[NFC7];
  int k = blockIdx.x;
  for (int i = threadIdx.x; i < NFC7; i += 256) hrow[i] = h7[(size_t)k * NFC7 + i];
  __syncthreads();
  int t = threadIdx.x;
  if (t < 2 * NCc) {
    int col = t % NCc;
    const float* w = (t < NCc) ? wc : wd;
    float acc = (t < NCc) ? bc[col] : bd[col];
    for (int c = 0; c < NFC7; ++c) acc += hrow[c] * w[(size_t)c * NCc + col];
    if (t < NCc) cl[(size_t)k * NCc + col] = acc;
    else         dl[(size_t)k * NCc + col] = acc;
  }
}

// softmax over axis=0 (2048 regions) per class column
__global__ __launch_bounds__(256)
void softmaxc_kernel(const float* __restrict__ cl, float* __restrict__ cn)
{
  __shared__ float red[256];
  int j = blockIdx.x;
  int t = threadIdx.x;
  float v[8];
  float m = -3.4e38f;
#pragma unroll
  for (int i = 0; i < 8; ++i) {
    v[i] = cl[(size_t)(t + 256 * i) * NCc + j];
    m = fmaxf(m, v[i]);
  }
  red[t] = m; __syncthreads();
  for (int s = 128; s > 0; s >>= 1) { if (t < s) red[t] = fmaxf(red[t], red[t + s]); __syncthreads(); }
  m = red[0]; __syncthreads();
  float sum = 0.0f;
#pragma unroll
  for (int i = 0; i < 8; ++i) { v[i] = __expf(v[i] - m); sum += v[i]; }
  red[t] = sum; __syncthreads();
  for (int s = 128; s > 0; s >>= 1) { if (t < s) red[t] += red[t + s]; __syncthreads(); }
  float inv = 1.0f / red[0];
#pragma unroll
  for (int i = 0; i < 8; ++i) cn[(size_t)(t + 256 * i) * NCc + j] = v[i] * inv;
}

// softmax over axis=1 (classes) fused with scores = c * d
__global__ void scores_kernel(const float* __restrict__ dl, const float* __restrict__ cn,
                              float* __restrict__ scores)
{
  int k = blockIdx.x * blockDim.x + threadIdx.x;
  if (k >= Kk) return;
  float row[NCc];
  float m = -3.4e38f;
#pragma unroll
  for (int j = 0; j < NCc; ++j) { row[j] = dl[(size_t)k * NCc + j]; m = fmaxf(m, row[j]); }
  float sum = 0.0f;
#pragma unroll
  for (int j = 0; j < NCc; ++j) { row[j] = __expf(row[j] - m); sum += row[j]; }
  float inv = 1.0f / sum;
#pragma unroll
  for (int j = 0; j < NCc; ++j)
    scores[(size_t)k * NCc + j] = cn[(size_t)k * NCc + j] * row[j] * inv;
}

// output = clip(sum over regions of scores, 0, 1)
__global__ __launch_bounds__(256)
void reduce_kernel(const float* __restrict__ scores, float* __restrict__ out)
{
  __shared__ float red[256];
  int t = threadIdx.x;
  float acc[NCc];
#pragma unroll
  for (int j = 0; j < NCc; ++j) acc[j] = 0.0f;
  for (int k = t; k < Kk; k += 256)
#pragma unroll
    for (int j = 0; j < NCc; ++j) acc[j] += scores[(size_t)k * NCc + j];
  for (int j = 0; j < NCc; ++j) {
    red[t] = acc[j]; __syncthreads();
    for (int s = 128; s > 0; s >>= 1) { if (t < s) red[t] += red[t + s]; __syncthreads(); }
    if (t == 0) out[j] = fminf(fmaxf(red[0], 0.0f), 1.0f);
    __syncthreads();
  }
}

// ---------------------------------------------------------------------------
extern "C" void kernel_launch(void* const* d_in, const int* in_sizes, int n_in,
                              void* d_out, int out_size, void* d_ws, size_t ws_size,
                              hipStream_t stream)
{
  const float* features = (const float*)d_in[0];
  // d_in[1] = labels (unused by forward outputs)
  const float* rpn_w  = (const float*)d_in[2];
  const float* rpn_b  = (const float*)d_in[3];
  const float* cls_w  = (const float*)d_in[4];
  const float* cls_b  = (const float*)d_in[5];
  const float* reg_w  = (const float*)d_in[6];
  const float* reg_b  = (const float*)d_in[7];
  const float* fc6_w  = (const float*)d_in[8];
  const float* fc6_b  = (const float*)d_in[9];
  const float* fc7_w  = (const float*)d_in[10];
  const float* fc7_b  = (const float*)d_in[11];
  const float* fc8c_w = (const float*)d_in[12];
  const float* fc8c_b = (const float*)d_in[13];
  const float* fc8d_w = (const float*)d_in[14];
  const float* fc8d_b = (const float*)d_in[15];

  char* ws = (char*)d_ws;
  size_t off = 0;
  auto take = [&](size_t bytes) -> char* {
    char* p = ws + off;
    off += (bytes + 255) & ~(size_t)255;
    return p;
  };

  unsigned short* A0   = (unsigned short*)take((size_t)MCONV * KCONV * 2);  // im2col bf16
  float*          X    = (float*)take((size_t)MCONV * NCONV * 4);           // rpn relu out
  float*          OBJ  = (float*)take((size_t)NANCH * 4);
  float*          REG4 = (float*)take((size_t)NANCH * 4 * 4);
  int*            IDX  = (int*)take((size_t)Kk * 4);
  float*          BOX  = (float*)take((size_t)Kk * 4 * 4);
  unsigned short* POOL = (unsigned short*)take((size_t)Kk * DINd * 2);      // bf16
  unsigned short* H6   = (unsigned short*)take((size_t)Kk * NFC6 * 2);      // bf16
  float*          H7   = (float*)take((size_t)Kk * NFC7 * 4);
  float*          CL   = (float*)take((size_t)Kk * NCc * 4);
  float*          DL   = (float*)take((size_t)Kk * NCc * 4);
  float*          CN   = (float*)take((size_t)Kk * NCc * 4);

  float* out        = (float*)d_out;
  float* out_scores = out + NCc;              // [2048*20]
  float* out_boxes  = out + NCc + Kk * NCc;   // [2048*4]

  // 1) im2col (features -> bf16 patches, zero-padded rows)
  {
    size_t n = (size_t)MCONV * KCONV;
    im2col_kernel<<<dim3((unsigned)((n + 255) / 256)), dim3(256), 0, stream>>>(features, A0);
  }
  // 2) RPN 2x2 conv as WMMA GEMM + bias + relu  (M=2560,K=2048,N=512)
  wmma_gemm<0><<<dim3(NCONV / 128, MCONV / 256), dim3(256), 0, stream>>>(
      A0, rpn_w, rpn_b, X, MCONV, NCONV, KCONV);
  // 3) 1x1 cls/reg heads
  clsreg_kernel<<<dim3(NPIX), dim3(64), 0, stream>>>(X, cls_w, cls_b, reg_w, reg_b, OBJ, REG4);
  // 4) exact top-2048
  topk_kernel<<<dim3((NANCH + 255) / 256), dim3(256), 0, stream>>>(OBJ, IDX);
  // 5) box decode
  boxes_kernel<<<dim3((Kk + 255) / 256), dim3(256), 0, stream>>>(IDX, REG4, BOX, out_boxes);
  // 6) ROI-align -> bf16 pooled features
  roialign_kernel<<<dim3(9, Kk), dim3(128), 0, stream>>>(features, BOX, POOL);
  // 7) fc6 WMMA GEMM + relu -> bf16  (M=2048,K=4608,N=9216)
  wmma_gemm<1><<<dim3(NFC6 / 128, Kk / 256), dim3(256), 0, stream>>>(
      POOL, fc6_w, fc6_b, H6, Kk, NFC6, DINd);
  // 8) fc7 WMMA GEMM + relu -> f32   (M=2048,K=9216,N=4096)
  wmma_gemm<0><<<dim3(NFC7 / 128, Kk / 256), dim3(256), 0, stream>>>(
      H6, fc7_w, fc7_b, H7, Kk, NFC7, NFC6);
  // 9) fc8c/fc8d logits
  fc8_kernel<<<dim3(Kk), dim3(256), 0, stream>>>(H7, fc8c_w, fc8c_b, fc8d_w, fc8d_b, CL, DL);
  // 10) softmax over regions (axis=0)
  softmaxc_kernel<<<dim3(NCc), dim3(256), 0, stream>>>(CL, CN);
  // 11) softmax over classes (axis=1) fused with scores = c*d
  scores_kernel<<<dim3((Kk + 255) / 256), dim3(256), 0, stream>>>(DL, CN, out_scores);
  // 12) clipped sum over regions
  reduce_kernel<<<dim3(1), dim3(256), 0, stream>>>(out_scores, out);
}
